// GCN_11046655885836
// MI455X (gfx1250) — matrix operand order repaired and checked by reading, not compiled, not run on hardware
//
#include <hip/hip_runtime.h>

// ---------------- WMMA vector types (CDNA5 / gfx1250, wave32) ----------------
typedef __attribute__((ext_vector_type(16))) __bf16 v16bf;
typedef __attribute__((ext_vector_type(8)))  float  v8f;

// ---------------- small helper kernels ----------------

// Split W (K x N, row-major fp32) into transposed bf16 hi/lo (N x K each):
// W[k,n] = hi[n,k] + lo[n,k]  (to bf16x2 precision)
__global__ void k_split_w(const float* __restrict__ W,
                          __bf16* __restrict__ hi, __bf16* __restrict__ lo,
                          int K, int N) {
    int idx = blockIdx.x * blockDim.x + threadIdx.x;
    if (idx >= K * N) return;
    int k = idx / N;
    int n = idx - k * N;
    float v = W[idx];
    __bf16 h = (__bf16)v;
    hi[(size_t)n * K + k] = h;
    lo[(size_t)n * K + k] = (__bf16)(v - (float)h);
}

__global__ void k_deg_init(float* __restrict__ deg, int n) {
    int i = blockIdx.x * blockDim.x + threadIdx.x;
    if (i < n) deg[i] = 1.0f;   // self-loop
}

__global__ void k_deg_count(const long long* __restrict__ dst,
                            float* __restrict__ deg, int e) {
    int i = blockIdx.x * blockDim.x + threadIdx.x;
    if (i < e) unsafeAtomicAdd(&deg[(int)dst[i]], 1.0f);
}

__global__ void k_deg_fin(float* __restrict__ deg, int n) {
    int i = blockIdx.x * blockDim.x + threadIdx.x;
    if (i < n) deg[i] = rsqrtf(deg[i]);   // deg -> dinv in place
}

// out[i,f] = bias[f] + h[i,f] * dinv[i]^2   (bias + self-loop message)
__global__ void k_init_out(const float* __restrict__ h,
                           const float* __restrict__ dinv,
                           const float* __restrict__ bias,
                           float* __restrict__ out, int n, int F) {
    int idx = blockIdx.x * blockDim.x + threadIdx.x;
    if (idx >= n * F) return;
    int i = idx / F;
    int f = idx - i * F;
    float di = dinv[i];
    out[idx] = bias[f] + h[idx] * di * di;
}

// One wave32 per edge: out[dst] += h[src] * dinv[src]*dinv[dst]
__global__ void k_scatter(const long long* __restrict__ src,
                          const long long* __restrict__ dst,
                          const float* __restrict__ h,
                          const float* __restrict__ dinv,
                          float* __restrict__ out, int e, int F) {
    int w = (int)((blockIdx.x * blockDim.x + threadIdx.x) >> 5);
    if (w >= e) return;
    int lane = threadIdx.x & 31;
    int s = (int)src[w];
    int d = (int)dst[w];
    float nrm = dinv[s] * dinv[d];
    int fpl = F >> 5;                       // 4 (F=128) or 2 (F=64)
    const float* hp = h + (size_t)s * F + lane * fpl;
    float* op = out + (size_t)d * F + lane * fpl;
#pragma unroll 4
    for (int j = 0; j < fpl; ++j)
        unsafeAtomicAdd(op + j, hp[j] * nrm);
}

// ---------------- WMMA GEMM: H[M,F] = act(A[M,K]) @ (Bh+Bl)[F,K]^T ----------
// Block = 128 threads = 4 waves; wave w handles output tile
// rows [bx*16, bx*16+16), cols [(by*4+w)*16, +16).
// fp32 A is split in-register into bf16 hi/lo; 3 WMMAs per 32-wide K step.
__global__ void k_gemm(const float* __restrict__ A,
                       const __bf16* __restrict__ Bh,
                       const __bf16* __restrict__ Bl,
                       float* __restrict__ H,
                       int K, int F, int relu_in) {
    const int lane = threadIdx.x & 31;
    const int wv   = threadIdx.x >> 5;
    const int m0   = blockIdx.x * 16;
    const int n0   = (blockIdx.y * 4 + wv) * 16;
    const int l15  = lane & 15;
    const int half = lane >> 4;

    // A layout (16-bit A 16x32): lane L, row = L&15;
    //   half 0 -> K offsets {0..15}, half 1 -> {16..31}
    const float* arow = A + (size_t)(m0 + l15) * K + half * 16;
    // B layout (16-bit B 32x16): lane L, col = L&15; K = half*16 + 0..15 (contig)
    const __bf16* bhp = Bh + (size_t)(n0 + l15) * K + half * 16;
    const __bf16* blp = Bl + (size_t)(n0 + l15) * K + half * 16;

    v8f c = {};
    for (int kk = 0; kk < K; kk += 32) {
        float4 f0 = *(const float4*)(arow + kk);
        float4 f1 = *(const float4*)(arow + kk + 4);
        float4 f2 = *(const float4*)(arow + kk + 8);
        float4 f3 = *(const float4*)(arow + kk + 12);
        float fa[16] = {f0.x, f0.y, f0.z, f0.w, f1.x, f1.y, f1.z, f1.w,
                        f2.x, f2.y, f2.z, f2.w, f3.x, f3.y, f3.z, f3.w};
        v16bf ah, al;
#pragma unroll
        for (int j = 0; j < 16; ++j) {
            float v = fa[j];
            if (relu_in) v = fmaxf(v, 0.0f);
            __bf16 hb = (__bf16)v;
            ah[j] = hb;
            al[j] = (__bf16)(v - (float)hb);
        }
        v16bf vbh = *(const v16bf*)(bhp + kk);   // 32B contiguous
        v16bf vbl = *(const v16bf*)(blp + kk);

        c = __builtin_amdgcn_wmma_f32_16x16x32_bf16(false, ah, false, vbh,
                                                    (short)0, c, false, false);
        c = __builtin_amdgcn_wmma_f32_16x16x32_bf16(false, al, false, vbh,
                                                    (short)0, c, false, false);
        c = __builtin_amdgcn_wmma_f32_16x16x32_bf16(false, ah, false, vbl,
                                                    (short)0, c, false, false);
    }

    // C/D layout: vgpr r -> row m0 + r + half*8, col n0 + (lane&15)
#pragma unroll
    for (int r = 0; r < 8; ++r)
        H[(size_t)(m0 + r + half * 8) * F + (n0 + l15)] = c[r];
}

// ---------------- launcher ----------------
extern "C" void kernel_launch(void* const* d_in, const int* in_sizes, int n_in,
                              void* d_out, int out_size, void* d_ws, size_t ws_size,
                              hipStream_t stream) {
    (void)n_in; (void)out_size; (void)ws_size;
    const float*     x  = (const float*)d_in[0];
    const long long* ei = (const long long*)d_in[1];   // (2, E) int64
    const float*     W1 = (const float*)d_in[2];
    const float*     b1 = (const float*)d_in[3];
    const float*     W2 = (const float*)d_in[4];
    const float*     b2 = (const float*)d_in[5];
    float*           out = (float*)d_out;

    const int K1 = 512, F1 = 128, F2 = 64;
    const int N = in_sizes[0] / K1;     // 50000 (divisible by 16)
    const int E = in_sizes[1] / 2;      // 1,600,000

    // carve workspace (256B aligned chunks)
    char* p = (char*)d_ws;
    auto carve = [&](size_t bytes) -> char* {
        char* q = p;
        p += (bytes + 255) & ~(size_t)255;
        return q;
    };
    float*  h    = (float*)carve((size_t)N * F1 * sizeof(float)); // h1, later h2
    float*  o1   = (float*)carve((size_t)N * F1 * sizeof(float)); // layer-1 output
    float*  dinv = (float*)carve((size_t)N * sizeof(float));
    __bf16* w1h  = (__bf16*)carve((size_t)K1 * F1 * sizeof(__bf16));
    __bf16* w1l  = (__bf16*)carve((size_t)K1 * F1 * sizeof(__bf16));
    __bf16* w2h  = (__bf16*)carve((size_t)F1 * F2 * sizeof(__bf16));
    __bf16* w2l  = (__bf16*)carve((size_t)F1 * F2 * sizeof(__bf16));

    // weight split/transpose to bf16 hi/lo
    k_split_w<<<(K1 * F1 + 255) / 256, 256, 0, stream>>>(W1, w1h, w1l, K1, F1);
    k_split_w<<<(F1 * F2 + 255) / 256, 256, 0, stream>>>(W2, w2h, w2l, F1, F2);

    // degree -> dinv (deg includes self-loop)
    k_deg_init<<<(N + 255) / 256, 256, 0, stream>>>(dinv, N);
    k_deg_count<<<(E + 255) / 256, 256, 0, stream>>>(ei + E, dinv, E);
    k_deg_fin<<<(N + 255) / 256, 256, 0, stream>>>(dinv, N);

    // ---- layer 1: h = x @ W1 ; o1 = scatter(h) + b1 ----
    k_gemm<<<dim3(N / 16, F1 / 64), 128, 0, stream>>>(x, w1h, w1l, h, K1, F1, 0);
    k_init_out<<<(N * F1 + 255) / 256, 256, 0, stream>>>(h, dinv, b1, o1, N, F1);
    k_scatter<<<(E + 7) / 8, 256, 0, stream>>>(ei, ei + E, h, dinv, o1, E, F1);

    // ---- layer 2: h2 = relu(o1) @ W2 ; out = scatter(h2) + b2 ----
    k_gemm<<<dim3(N / 16, F2 / 64), 128, 0, stream>>>(o1, w2h, w2l, h, F1, F2, 1);
    k_init_out<<<(N * F2 + 255) / 256, 256, 0, stream>>>(h, dinv, b2, out, N, F2);
    k_scatter<<<(E + 7) / 8, 256, 0, stream>>>(ei, ei + E, h, dinv, out, E, F2);
}
